// TransformerModel_27599459844625
// MI455X (gfx1250) — matrix operand order repaired and checked by reading, not compiled, not run on hardware
//
#include <hip/hip_runtime.h>
#include <hip/hip_bf16.h>

// ---------------------------------------------------------------------------
// Model constants
// ---------------------------------------------------------------------------
#define EMBD 384
#define FFN  1536
#define SEQ  200        // S = T_IN + 1
#define T_IN 199
#define NB   128        // batch
#define NH   6
#define HS   64
#define NL   6
#define ROWS (NB*SEQ)   // 25600, multiple of 16
#define HROWS (NB*T_IN) // 25472, multiple of 16
#define SQ   208        // padded query rows (13 tiles)
#define SKP  224        // padded key rows   (14 tiles, multiple of 32)
#define QT   13
#define KT   14

typedef __attribute__((ext_vector_type(16))) __bf16        v16bf;
typedef __attribute__((ext_vector_type(8)))  float         v8f;
typedef __attribute__((ext_vector_type(4)))  unsigned int  v4u;
typedef __attribute__((ext_vector_type(8)))  unsigned int  v8u;

union AFrag { v16bf bf; v4u u[2]; };
union BFrag { v16bf bf; v8u u; };

__device__ __forceinline__ unsigned short f2bf(float f) {
  unsigned int u = __float_as_uint(f);
  u += 0x7FFFu + ((u >> 16) & 1u);       // round-to-nearest-even
  return (unsigned short)(u >> 16);
}

// ---------------------------------------------------------------------------
// Weight packing: B-fragment layout.
// packed[((nt*KC+kc)*32 + lane)*16 + i] = bf16( W[kc*32 + (lane>>4)*16 + i][nt*16 + (lane&15)] )
// so each lane's 16 bf16 (32 B) are contiguous.
// ---------------------------------------------------------------------------
__global__ void pack_w_k(const float* __restrict__ src, unsigned short* __restrict__ dst,
                         int K, int N) {
  long idx = (long)blockIdx.x * blockDim.x + threadIdx.x;
  long total = (long)K * N;
  if (idx >= total) return;
  int  i    = (int)(idx & 15);
  int  lane = (int)((idx >> 4) & 31);
  long tile = idx >> 9;
  int  KC   = K >> 5;
  int  kc   = (int)(tile % KC);
  int  nt   = (int)(tile / KC);
  int  k    = kc*32 + (lane >> 4)*16 + i;
  int  n    = nt*16 + (lane & 15);
  dst[idx]  = f2bf(src[(long)k * N + n]);
}

// Combined QKV weight for one layer: K=384, N=1152; col n -> (sec, h, d)
__global__ void pack_qkv_k(const float* __restrict__ Wq, const float* __restrict__ Wk,
                           const float* __restrict__ Wv, unsigned short* __restrict__ dst) {
  long idx = (long)blockIdx.x * blockDim.x + threadIdx.x;
  long total = (long)EMBD * (3*EMBD);
  if (idx >= total) return;
  int  i    = (int)(idx & 15);
  int  lane = (int)((idx >> 4) & 31);
  long tile = idx >> 9;
  int  KC   = EMBD >> 5;                 // 12
  int  kc   = (int)(tile % KC);
  int  nt   = (int)(tile / KC);
  int  k    = kc*32 + (lane >> 4)*16 + i;      // c in [0,384)
  int  n    = nt*16 + (lane & 15);             // [0,1152)
  int  sec  = n / EMBD;
  int  nn   = n % EMBD;
  int  h    = nn >> 6, d = nn & 63;
  const float* src = (sec == 0) ? Wq : (sec == 1) ? Wk : Wv;
  dst[idx] = f2bf(src[((long)h*EMBD + k)*HS + d]);
}

// ---------------------------------------------------------------------------
// Embedding
// ---------------------------------------------------------------------------
__global__ void embed_k(const float* __restrict__ x, const int* __restrict__ cls_idx,
                        const float* __restrict__ Wstroke, const float* __restrict__ bstroke,
                        const float* __restrict__ pen_emb, const float* __restrict__ pos_emb,
                        const float* __restrict__ cls_emb, float* __restrict__ h) {
  long idx = (long)blockIdx.x * blockDim.x + threadIdx.x;
  if (idx >= (long)ROWS * EMBD) return;
  int  e   = (int)(idx % EMBD);
  long row = idx / EMBD;
  int  b   = (int)(row / SEQ);
  int  s   = (int)(row % SEQ);
  float v;
  if (s == 0) {
    v = cls_emb[(long)cls_idx[b]*EMBD + e];
  } else {
    int t = s - 1;
    const float* xr = x + ((long)b*T_IN + t)*3;
    int pi = (int)xr[2]; pi = pi < 0 ? 0 : (pi > 2 ? 2 : pi);
    v = xr[0]*Wstroke[e] + xr[1]*Wstroke[EMBD+e] + bstroke[e]
      + pen_emb[(long)pi*EMBD + e] + pos_emb[(long)t*EMBD + e];
  }
  h[idx] = v;
}

// ---------------------------------------------------------------------------
// LayerNorm over 384, one block (128 threads) per row. bf16 and/or f32 output.
// ---------------------------------------------------------------------------
__global__ __launch_bounds__(128)
void layernorm_k(const float* __restrict__ x, const float* __restrict__ gain,
                 const float* __restrict__ bias, unsigned short* __restrict__ outB,
                 float* __restrict__ outF) {
  long row = blockIdx.x;
  const float* xr = x + row*EMBD;
  __shared__ float red[128];
  int t = threadIdx.x;
  float v0 = xr[t], v1 = xr[t+128], v2 = xr[t+256];
  red[t] = v0 + v1 + v2;
  __syncthreads();
  for (int o = 64; o > 0; o >>= 1) { if (t < o) red[t] += red[t+o]; __syncthreads(); }
  float mu = red[0] * (1.0f/EMBD);
  __syncthreads();
  float d0 = v0-mu, d1 = v1-mu, d2 = v2-mu;
  red[t] = d0*d0 + d1*d1 + d2*d2;
  __syncthreads();
  for (int o = 64; o > 0; o >>= 1) { if (t < o) red[t] += red[t+o]; __syncthreads(); }
  float rstd = rsqrtf(red[0]*(1.0f/EMBD) + 1e-5f);
  float y0 = d0*rstd*gain[t]     + bias[t];
  float y1 = d1*rstd*gain[t+128] + bias[t+128];
  float y2 = d2*rstd*gain[t+256] + bias[t+256];
  if (outB) {
    outB[row*EMBD + t]     = f2bf(y0);
    outB[row*EMBD + t+128] = f2bf(y1);
    outB[row*EMBD + t+256] = f2bf(y2);
  }
  if (outF) {
    outF[row*EMBD + t]     = y0;
    outF[row*EMBD + t+128] = y1;
    outF[row*EMBD + t+256] = y2;
  }
}

// ---------------------------------------------------------------------------
// WMMA GEMM: one wave computes a 16x64 tile (4 N-tiles, A-fragment reused 4x):
//   out = A(bf16, MxK row-major) x Bp(packed bf16) [+bias][relu][+residual]
// ---------------------------------------------------------------------------
__global__ __launch_bounds__(32)
void gemm_bf16_k(const unsigned short* __restrict__ A, const unsigned short* __restrict__ Bp,
                 const float* __restrict__ bias, const float* __restrict__ residual,
                 float* __restrict__ outF, unsigned short* __restrict__ outB,
                 int M, int K, int N, int relu) {
  const int lane = threadIdx.x & 31;
  const int g    = lane >> 4;
  const int mt   = blockIdx.x;
  const int nt0  = blockIdx.y * 4;          // 4 consecutive N-tiles per wave
  const int KC   = K >> 5;
  const unsigned short* Arow = A + (size_t)(mt*16 + (lane & 15)) * K;
  v8f c0 = {}, c1 = {}, c2 = {}, c3 = {};
  for (int kc = 0; kc < KC; ++kc) {
    AFrag a;
    a.u[0] = *(const v4u*)(Arow + kc*32 + g*16);
    a.u[1] = *(const v4u*)(Arow + kc*32 + g*16 + 8);
    const unsigned short* bbase = Bp + ((size_t)kc*32 + lane) * 16;
    BFrag b0, b1, b2, b3;
    b0.u = *(const v8u*)(bbase + (size_t)(nt0+0)*KC*512);
    b1.u = *(const v8u*)(bbase + (size_t)(nt0+1)*KC*512);
    b2.u = *(const v8u*)(bbase + (size_t)(nt0+2)*KC*512);
    b3.u = *(const v8u*)(bbase + (size_t)(nt0+3)*KC*512);
    c0 = __builtin_amdgcn_wmma_f32_16x16x32_bf16(false, a.bf, false, b0.bf, (short)0, c0, false, false);
    c1 = __builtin_amdgcn_wmma_f32_16x16x32_bf16(false, a.bf, false, b1.bf, (short)0, c1, false, false);
    c2 = __builtin_amdgcn_wmma_f32_16x16x32_bf16(false, a.bf, false, b2.bf, (short)0, c2, false, false);
    c3 = __builtin_amdgcn_wmma_f32_16x16x32_bf16(false, a.bf, false, b3.bf, (short)0, c3, false, false);
  }
  v8f acc[4] = { c0, c1, c2, c3 };
#pragma unroll
  for (int tt = 0; tt < 4; ++tt) {
    const int col = (nt0 + tt)*16 + (lane & 15);
    const float bv = bias ? bias[col] : 0.0f;
#pragma unroll
    for (int rr = 0; rr < 8; ++rr) {
      int row = mt*16 + g*8 + rr;
      float v = acc[tt][rr] + bv;
      if (relu) v = v > 0.0f ? v : 0.0f;
      size_t off = (size_t)row * N + col;
      if (residual) v += residual[off];
      if (outF) outF[off] = v;
      if (outB) outB[off] = f2bf(v);
    }
  }
}

// ---------------------------------------------------------------------------
// Reshape QKV GEMM output (ROWS x 1152 bf16) into attention-friendly layouts:
//   Qb[bh][208][64] (zero-padded), Kb[bh][224][64] (zero-padded),
//   Vt[bh][64][224] (transposed, zero-padded)
// ---------------------------------------------------------------------------
__global__ void reshape_qkv_k(const unsigned short* __restrict__ qkv,
                              unsigned short* __restrict__ Qb,
                              unsigned short* __restrict__ Kb,
                              unsigned short* __restrict__ Vt) {
  long idx = (long)blockIdx.x * blockDim.x + threadIdx.x;
  long total = (long)NB*NH * SKP * HS;
  if (idx >= total) return;
  int d  = (int)(idx & 63);
  int s  = (int)((idx >> 6) % SKP);
  int bh = (int)(idx / (64*SKP));
  int b  = bh / NH, hh = bh % NH;
  unsigned short q = 0, k = 0, v = 0;
  if (s < SEQ) {
    const unsigned short* r = qkv + ((size_t)b*SEQ + s) * (3*EMBD);
    q = r[hh*HS + d];
    k = r[EMBD   + hh*HS + d];
    v = r[2*EMBD + hh*HS + d];
  }
  if (s < SQ) Qb[((size_t)bh*SQ + s)*HS + d] = q;
  Kb[((size_t)bh*SKP + s)*HS + d] = k;
  Vt[((size_t)bh*HS + d)*SKP + s] = v;
}

// ---------------------------------------------------------------------------
// Attention: one wave per (b,h, 16-query tile). WMMA for QK^T and P·V,
// softmax through LDS (scale = EMBD^-0.5, causal mask).
// ---------------------------------------------------------------------------
__global__ __launch_bounds__(32)
void attn_k(const unsigned short* __restrict__ Qb, const unsigned short* __restrict__ Kb,
            const unsigned short* __restrict__ Vt, unsigned short* __restrict__ obf) {
  const int bh   = blockIdx.x;     // 0..767
  const int qt   = blockIdx.y;     // 0..12
  const int b    = bh / NH, hh = bh % NH;
  const int lane = threadIdx.x & 31;
  const int g    = lane >> 4;
  const unsigned short* Q  = Qb + (size_t)bh * SQ  * HS;
  const unsigned short* Kp = Kb + (size_t)bh * SKP * HS;
  const unsigned short* V  = Vt + (size_t)bh * HS  * SKP;

  __shared__ __align__(16) float          sc[16*SKP];
  __shared__ __align__(16) unsigned short pb[16*SKP];

  // Q fragments (K=64 -> 2 chunks of 32)
  AFrag aq[2];
  {
    const unsigned short* qr = Q + (size_t)(qt*16 + (lane & 15)) * HS;
#pragma unroll
    for (int kc = 0; kc < 2; ++kc) {
      aq[kc].u[0] = *(const v4u*)(qr + kc*32 + g*16);
      aq[kc].u[1] = *(const v4u*)(qr + kc*32 + g*16 + 8);
    }
  }
  const float scale = 0.05103103630798288f;   // 384^-0.5

  for (int kt = 0; kt <= qt; ++kt) {
    v8f c = {};
#pragma unroll
    for (int kc = 0; kc < 2; ++kc) {
      BFrag bk;
      bk.u = *(const v8u*)(Kp + (size_t)(kt*16 + (lane & 15)) * HS + kc*32 + g*16);
      c = __builtin_amdgcn_wmma_f32_16x16x32_bf16(false, aq[kc].bf, false, bk.bf,
                                                  (short)0, c, false, false);
    }
    int key = kt*16 + (lane & 15);
#pragma unroll
    for (int rr = 0; rr < 8; ++rr) {
      int rl   = g*8 + rr;
      int qrow = qt*16 + rl;
      sc[rl*SKP + key] = (key <= qrow) ? c[rr]*scale : -1e30f;
    }
  }
  for (int kt = qt+1; kt < KT; ++kt) {
    int key = kt*16 + (lane & 15);
#pragma unroll
    for (int rr = 0; rr < 8; ++rr) sc[(g*8+rr)*SKP + key] = -1e30f;
  }
  __syncthreads();

  if (lane < 16) {
    int rl = lane;
    int qrow = qt*16 + rl;
    if (qrow < SEQ) {
      float m = -1e30f;
      for (int j = 0; j < SKP; ++j) m = fmaxf(m, sc[rl*SKP + j]);
      float s = 0.0f;
      for (int j = 0; j < SKP; ++j) { float e = __expf(sc[rl*SKP + j] - m); sc[rl*SKP + j] = e; s += e; }
      float inv = 1.0f / s;
      for (int j = 0; j < SKP; ++j) pb[rl*SKP + j] = f2bf(sc[rl*SKP + j] * inv);
    } else {
      for (int j = 0; j < SKP; ++j) pb[rl*SKP + j] = 0;
    }
  }
  __syncthreads();

  // O = P (16 x 224) @ V (224 x 64): P-fragment loaded once per K-chunk,
  // reused across all 4 output-column tiles.
  const unsigned short* pr = pb + (size_t)(lane & 15) * SKP;
  v8f o0 = {}, o1 = {}, o2 = {}, o3 = {};
  for (int kc = 0; kc < SKP/32; ++kc) {
    AFrag ap;
    ap.u[0] = *(const v4u*)(pr + kc*32 + g*16);
    ap.u[1] = *(const v4u*)(pr + kc*32 + g*16 + 8);
    const unsigned short* vbase = V + (size_t)(lane & 15)*SKP + kc*32 + g*16;
    BFrag b0, b1, b2, b3;
    b0.u = *(const v8u*)(vbase + (size_t)0*16*SKP);
    b1.u = *(const v8u*)(vbase + (size_t)1*16*SKP);
    b2.u = *(const v8u*)(vbase + (size_t)2*16*SKP);
    b3.u = *(const v8u*)(vbase + (size_t)3*16*SKP);
    o0 = __builtin_amdgcn_wmma_f32_16x16x32_bf16(false, ap.bf, false, b0.bf, (short)0, o0, false, false);
    o1 = __builtin_amdgcn_wmma_f32_16x16x32_bf16(false, ap.bf, false, b1.bf, (short)0, o1, false, false);
    o2 = __builtin_amdgcn_wmma_f32_16x16x32_bf16(false, ap.bf, false, b2.bf, (short)0, o2, false, false);
    o3 = __builtin_amdgcn_wmma_f32_16x16x32_bf16(false, ap.bf, false, b3.bf, (short)0, o3, false, false);
  }
  v8f oacc[4] = { o0, o1, o2, o3 };
#pragma unroll
  for (int dt = 0; dt < 4; ++dt) {
#pragma unroll
    for (int rr = 0; rr < 8; ++rr) {
      int row = qt*16 + g*8 + rr;
      if (row < SEQ) {
        int col = hh*HS + dt*16 + (lane & 15);
        obf[((size_t)b*SEQ + row)*EMBD + col] = f2bf(oacc[dt][rr]);
      }
    }
  }
}

// ---------------------------------------------------------------------------
// MDN heads (small, fp32): pi(20) | mean(40) | tril(60->80 with exp diag) | pen(3)
// ---------------------------------------------------------------------------
__global__ __launch_bounds__(128)
void heads_k(const float* __restrict__ hf,
             const float* __restrict__ Wpi,   const float* __restrict__ bpi,
             const float* __restrict__ Wmean, const float* __restrict__ bmean,
             const float* __restrict__ Wtril, const float* __restrict__ btril,
             const float* __restrict__ Wpen,  const float* __restrict__ bpen,
             float* __restrict__ out) {
  long idx = (long)blockIdx.x * blockDim.x + threadIdx.x;
  const long rows = HROWS;
  if (idx >= rows * 123) return;
  int  c   = (int)(idx % 123);
  long row = idx / 123;
  int  b   = (int)(row / T_IN);
  int  t   = (int)(row % T_IN);
  const float* hr = hf + ((size_t)b*SEQ + t + 1) * EMBD;

  const float* W; const float* bias; int n, cc;
  if      (c < 20)  { W = Wpi;   bias = bpi;   n = 20; cc = c;      }
  else if (c < 60)  { W = Wmean; bias = bmean; n = 40; cc = c - 20; }
  else if (c < 120) { W = Wtril; bias = btril; n = 60; cc = c - 60; }
  else              { W = Wpen;  bias = bpen;  n = 3;  cc = c - 120;}

  float acc = bias[cc];
  for (int e = 0; e < EMBD; ++e) acc += hr[e] * W[(size_t)e*n + cc];

  const long PI_OFF = 0, MEAN_OFF = rows*20, TRIL_OFF = MEAN_OFF + rows*40,
             PEN_OFF = TRIL_OFF + rows*80;
  if      (c < 20)  out[PI_OFF   + row*20 + cc] = acc;
  else if (c < 60)  out[MEAN_OFF + row*40 + cc] = acc;
  else if (c < 120) {
    int k = cc / 3, w = cc % 3;
    float* base = out + TRIL_OFF + row*80 + (long)k*4;
    if      (w == 0) { base[0] = __expf(acc); base[1] = 0.0f; }
    else if (w == 1) { base[2] = acc; }
    else             { base[3] = __expf(acc); }
  }
  else out[PEN_OFF + row*3 + cc] = acc;
}

// ---------------------------------------------------------------------------
// Host launcher
// ---------------------------------------------------------------------------
extern "C" void kernel_launch(void* const* d_in, const int* in_sizes, int n_in,
                              void* d_out, int out_size, void* d_ws, size_t ws_size,
                              hipStream_t stream) {
  const float* x        = (const float*)d_in[0];
  const int*   cls_idx  = (const int*)  d_in[1];
  const float* Wstroke  = (const float*)d_in[2];
  const float* bstroke  = (const float*)d_in[3];
  const float* pen_emb  = (const float*)d_in[4];
  const float* pos_emb  = (const float*)d_in[5];
  const float* cls_emb  = (const float*)d_in[6];
  const float* Wq       = (const float*)d_in[7];
  const float* Wk       = (const float*)d_in[8];
  const float* Wv       = (const float*)d_in[9];
  const float* Wproj    = (const float*)d_in[10];
  const float* bproj    = (const float*)d_in[11];
  const float* ln1_g    = (const float*)d_in[12];
  const float* ln1_b    = (const float*)d_in[13];
  const float* ln2_g    = (const float*)d_in[14];
  const float* ln2_b    = (const float*)d_in[15];
  const float* W1       = (const float*)d_in[16];
  const float* b1       = (const float*)d_in[17];
  const float* W2       = (const float*)d_in[18];
  const float* b2       = (const float*)d_in[19];
  const float* lnf_g    = (const float*)d_in[20];
  const float* lnf_b    = (const float*)d_in[21];
  const float* Wpi      = (const float*)d_in[22];
  const float* bpi      = (const float*)d_in[23];
  const float* Wmean    = (const float*)d_in[24];
  const float* bmean    = (const float*)d_in[25];
  const float* Wtril    = (const float*)d_in[26];
  const float* btril    = (const float*)d_in[27];
  const float* Wpen     = (const float*)d_in[28];
  const float* bpen     = (const float*)d_in[29];
  float* out = (float*)d_out;

  // workspace carving (256B aligned)
  char* ws = (char*)d_ws;
  size_t off = 0;
  auto alloc = [&](size_t bytes) -> char* {
    char* p = ws + off;
    off = (off + bytes + 255) & ~(size_t)255;
    return p;
  };
  float*          h_f   = (float*)         alloc((size_t)ROWS*EMBD*4);
  unsigned short* ybf   = (unsigned short*)alloc((size_t)ROWS*EMBD*2);
  unsigned short* big   = (unsigned short*)alloc((size_t)ROWS*FFN*2);   // qkv/ffn1 out & lnf f32
  unsigned short* obf   = (unsigned short*)alloc((size_t)ROWS*EMBD*2);
  unsigned short* Qb    = (unsigned short*)alloc((size_t)NB*NH*SQ*HS*2);
  unsigned short* Kb    = (unsigned short*)alloc((size_t)NB*NH*SKP*HS*2);
  unsigned short* Vt    = (unsigned short*)alloc((size_t)NB*NH*HS*SKP*2);
  unsigned short* pwqkv = (unsigned short*)alloc((size_t)NL*EMBD*3*EMBD*2);
  unsigned short* pwprj = (unsigned short*)alloc((size_t)NL*EMBD*EMBD*2);
  unsigned short* pww1  = (unsigned short*)alloc((size_t)NL*EMBD*FFN*2);
  unsigned short* pww2  = (unsigned short*)alloc((size_t)NL*FFN*EMBD*2);
  (void)ws_size; (void)n_in; (void)in_sizes; (void)out_size;

  // -------- weight packing (every call; deterministic) --------
  for (int l = 0; l < NL; ++l) {
    long nqkv = (long)EMBD*3*EMBD;
    pack_qkv_k<<<(nqkv + 255)/256, 256, 0, stream>>>(
        Wq + (size_t)l*NH*EMBD*HS, Wk + (size_t)l*NH*EMBD*HS, Wv + (size_t)l*NH*EMBD*HS,
        pwqkv + (size_t)l*EMBD*3*EMBD);
    long nprj = (long)EMBD*EMBD;
    pack_w_k<<<(nprj + 255)/256, 256, 0, stream>>>(
        Wproj + (size_t)l*EMBD*EMBD, pwprj + (size_t)l*EMBD*EMBD, EMBD, EMBD);
    long nw1 = (long)EMBD*FFN;
    pack_w_k<<<(nw1 + 255)/256, 256, 0, stream>>>(
        W1 + (size_t)l*EMBD*FFN, pww1 + (size_t)l*EMBD*FFN, EMBD, FFN);
    long nw2 = (long)FFN*EMBD;
    pack_w_k<<<(nw2 + 255)/256, 256, 0, stream>>>(
        W2 + (size_t)l*FFN*EMBD, pww2 + (size_t)l*FFN*EMBD, FFN, EMBD);
  }

  // -------- embedding --------
  {
    long n = (long)ROWS*EMBD;
    embed_k<<<(n + 255)/256, 256, 0, stream>>>(x, cls_idx, Wstroke, bstroke,
                                               pen_emb, pos_emb, cls_emb, h_f);
  }

  // -------- transformer layers --------
  for (int l = 0; l < NL; ++l) {
    // LN1 -> ybf
    layernorm_k<<<ROWS, 128, 0, stream>>>(h_f, ln1_g + l*EMBD, ln1_b + l*EMBD, ybf, nullptr);
    // QKV gemm: (25600x384)x(384x1152) -> big (bf16)
    gemm_bf16_k<<<dim3(ROWS/16, (3*EMBD)/64), 32, 0, stream>>>(
        ybf, pwqkv + (size_t)l*EMBD*3*EMBD, nullptr, nullptr, nullptr, big,
        ROWS, EMBD, 3*EMBD, 0);
    // reshape to attention layouts
    {
      long n = (long)NB*NH*SKP*HS;
      reshape_qkv_k<<<(n + 255)/256, 256, 0, stream>>>(big, Qb, Kb, Vt);
    }
    // attention -> obf
    attn_k<<<dim3(NB*NH, QT), 32, 0, stream>>>(Qb, Kb, Vt, obf);
    // proj gemm + bias + residual -> h_f
    gemm_bf16_k<<<dim3(ROWS/16, EMBD/64), 32, 0, stream>>>(
        obf, pwprj + (size_t)l*EMBD*EMBD, bproj + l*EMBD, h_f, h_f, nullptr,
        ROWS, EMBD, EMBD, 0);
    // LN2 -> ybf
    layernorm_k<<<ROWS, 128, 0, stream>>>(h_f, ln2_g + l*EMBD, ln2_b + l*EMBD, ybf, nullptr);
    // FFN1 gemm + bias + relu -> big (bf16)
    gemm_bf16_k<<<dim3(ROWS/16, FFN/64), 32, 0, stream>>>(
        ybf, pww1 + (size_t)l*EMBD*FFN, b1 + l*FFN, nullptr, nullptr, big,
        ROWS, EMBD, FFN, 1);
    // FFN2 gemm + bias + residual -> h_f
    gemm_bf16_k<<<dim3(ROWS/16, EMBD/64), 32, 0, stream>>>(
        big, pww2 + (size_t)l*FFN*EMBD, b2 + l*EMBD, h_f, h_f, nullptr,
        ROWS, FFN, EMBD, 0);
  }

  // -------- final LN (f32 into big) + heads --------
  layernorm_k<<<ROWS, 128, 0, stream>>>(h_f, lnf_g, lnf_b, nullptr, (float*)big);
  {
    long n = (long)HROWS * 123;
    heads_k<<<(n + 127)/128, 128, 0, stream>>>((const float*)big,
        Wpi, bpi, Wmean, bmean, Wtril, btril, Wpen, bpen, out);
  }
}
